// PAtten_55525337202837
// MI455X (gfx1250) — matrix operand order repaired and checked by reading, not compiled, not run on hardware
//
#include <hip/hip_runtime.h>

// ---------------------------------------------------------------------------
// PAtten for MI455X (gfx1250, wave32, WMMA).
//  - Q convs (7x7, 11x11) as per-tap implicit GEMMs on v_wmma_f32_16x16x32_bf16
//    (weights pre-packed bf16 [tap][co][ci]; one padded image row staged in LDS
//     per kh, kw taps slide by pointer offset).
//  - Fused flash attention (bf16 WMMA, online softmax, att never materialized).
//  - K/V tiles staged via GLOBAL_LOAD_ASYNC_TO_LDS (ASYNCcnt) when available,
//    with s_wait_asynccnt before the barrier; VGPR-bounce fallback otherwise.
//  - bf16 WMMA GEMMs for the two 256x256 1x1 convs (BN+SiLU fused into c1).
//  - global_prefetch of next K/V tile in the attention key loop.
// ---------------------------------------------------------------------------

typedef __bf16 bf16;
typedef __attribute__((ext_vector_type(16))) __bf16 v16bf;
typedef __attribute__((ext_vector_type(8)))  float  v8f;
typedef __attribute__((ext_vector_type(4)))  int    v4i;

#if defined(__has_builtin)
#if __has_builtin(__builtin_amdgcn_global_load_async_to_lds_b128)
#define HAVE_ASYNC_LDS 1
#endif
#endif
#ifndef HAVE_ASYNC_LDS
#define HAVE_ASYNC_LDS 0
#endif

constexpr int B  = 8;
constexpr int C  = 256;
constexpr int CA = 64;
constexpr int H  = 64;
constexpr int W  = 64;
constexpr int HW = H * W;

__device__ __forceinline__ v8f wmma_bf16(v16bf a, v16bf b, v8f c) {
  // D = A(16x32 bf16) * B(32x16 bf16) + C(16x16 f32)
  return __builtin_amdgcn_wmma_f32_16x16x32_bf16(false, a, false, b,
                                                 (short)0, c, false, false);
}

// A-matrix fragment (16x32 bf16) from row-major [m][k] with leading dim ld.
__device__ __forceinline__ v16bf load_frag_a(const bf16* p, int ld) {
  int lane = threadIdx.x & 31;
  int g = lane >> 4;
  int m = lane & 15;
  v16bf r;
#pragma unroll
  for (int e = 0; e < 16; ++e) {
    int k = ((e >> 3) << 4) + (g << 3) + (e & 7);
    r[e] = p[m * ld + k];
  }
  return r;
}

// B-matrix fragment (32x16 bf16), element (k,n) at p[k*sk + n*sn].
__device__ __forceinline__ v16bf load_frag_b(const bf16* p, int sk, int sn) {
  int lane = threadIdx.x & 31;
  int g = lane >> 4;
  int n = lane & 15;
  v16bf r;
#pragma unroll
  for (int e = 0; e < 16; ++e) {
    int k = ((e >> 3) << 4) + (g << 3) + (e & 7);
    r[e] = p[k * sk + n * sn];
  }
  return r;
}

// ---------------------------------------------------------------------------
// Weight prepack: w[co][ci][kh][kw] f32 -> wb[tap][co][ci] bf16  (tap=kh*KS+kw)
// ---------------------------------------------------------------------------
template <int KS>
__global__ void __launch_bounds__(256) prepack_w_kernel(
    const float* __restrict__ w, bf16* __restrict__ wb) {
  int idx = blockIdx.x * 256 + threadIdx.x;
  if (idx >= 32 * CA * KS * KS) return;
  int ci  = idx & 63;
  int co  = (idx >> 6) & 31;
  int tap = idx >> 11;
  wb[idx] = (bf16)w[((size_t)co * CA + ci) * (KS * KS) + tap];
}

// ---------------------------------------------------------------------------
// Q convs as implicit GEMM: per tap  out(32co x 64cols) += Wtap * Xshift.
// Block = 128 threads (4 waves) handles one (b, h) image row.
// ---------------------------------------------------------------------------
template <int KS, int PAD>
__global__ void __launch_bounds__(128) conv_q_wmma_kernel(
    const float* __restrict__ x, const bf16* __restrict__ wb,
    const float* __restrict__ bias, bf16* __restrict__ qT, int cbase) {
  __shared__ bf16 sX[64 * 80];
  int b = blockIdx.y;
  int h = blockIdx.x;
  int wave = threadIdx.x >> 5;
  int lane = threadIdx.x & 31;
  int g = lane >> 4, ln = lane & 15;
  int mt = wave & 1;
  int nb = (wave >> 1) * 32;

  v8f zero = {0.f, 0.f, 0.f, 0.f, 0.f, 0.f, 0.f, 0.f};
  v8f acc[2] = {zero, zero};

  for (int kh = 0; kh < KS; ++kh) {
    int h2 = h + kh - PAD;             // uniform across block
    if (h2 < 0 || h2 >= H) continue;
    __syncthreads();                   // waves done reading previous sX
    for (int i = threadIdx.x; i < 64 * 80; i += 128) {
      int ci = i / 80, j = i - ci * 80;
      int w2 = j - PAD;
      float v = (w2 >= 0 && w2 < W)
                    ? x[((size_t)b * C + ci) * HW + h2 * W + w2] : 0.f;
      sX[i] = (bf16)v;
    }
    __syncthreads();
    for (int kw = 0; kw < KS; ++kw) {
      const bf16* wt = wb + ((size_t)(kh * KS + kw) * 32 + mt * 16) * CA;
      v16bf a0 = load_frag_a(wt, CA);        // ci 0..31
      v16bf a1 = load_frag_a(wt + 32, CA);   // ci 32..63
#pragma unroll
      for (int t = 0; t < 2; ++t) {
        const bf16* bp = sX + nb + t * 16 + kw;  // (k=ci, n): bp[ci*80 + n]
        acc[t] = wmma_bf16(a0, load_frag_b(bp, 80, 1), acc[t]);
        acc[t] = wmma_bf16(a1, load_frag_b(bp + 32 * 80, 80, 1), acc[t]);
      }
    }
  }

#pragma unroll
  for (int t = 0; t < 2; ++t) {
#pragma unroll
    for (int r = 0; r < 8; ++r) {
      int co = mt * 16 + r + 8 * g;
      int wc = nb + t * 16 + ln;
      int n  = h * W + wc;
      qT[((size_t)b * HW + n) * CA + cbase + co] = (bf16)(acc[t][r] + bias[co]);
    }
  }
}

// ---------------------------------------------------------------------------
// K/V 1x1 convs -> bf16 kT/vT[b][n][co]   (tiny: 0.5 GFLOP total, scalar VALU)
// ---------------------------------------------------------------------------
__global__ void __launch_bounds__(256) kv_kernel(
    const float* __restrict__ x,
    const float* __restrict__ w_k, const float* __restrict__ b_k,
    const float* __restrict__ w_v, const float* __restrict__ b_v,
    bf16* __restrict__ kT, bf16* __restrict__ vT) {
  int idx = blockIdx.x * 256 + threadIdx.x;   // over B*64*HW
  int n  = idx & (HW - 1);
  int t  = idx >> 12;
  int co = t & 63;
  int b  = t >> 6;
  const float* xb = x + (size_t)b * C * HW + n;
  float ak = b_k[co], av = b_v[co];
  for (int ci = 0; ci < CA; ++ci) {
    float xv = xb[(size_t)ci * HW];
    ak += w_k[co * CA + ci] * xv;
    av += w_v[co * CA + ci] * xv;
  }
  size_t o = ((size_t)b * HW + n) * CA + co;
  kT[o] = (bf16)ak;
  vT[o] = (bf16)av;
}

// ---------------------------------------------------------------------------
// Flash attention: per wave 16 query rows, key tiles of 32 staged in LDS.
// out1[b][c][n] f32 (c = value channel 0..63, n = query index)
// ---------------------------------------------------------------------------
#define ATT_WAVES 4
__global__ void __launch_bounds__(128) attn_kernel(
    const bf16* __restrict__ qT, const bf16* __restrict__ kT,
    const bf16* __restrict__ vT, float* __restrict__ out1) {
  __shared__ bf16 sK[32 * 64];
  __shared__ bf16 sV[32 * 64];
  __shared__ float sS[ATT_WAVES][16 * 32];
  __shared__ bf16  sP[ATT_WAVES][16 * 32];
  __shared__ float sM[ATT_WAVES][16];
  __shared__ float sL[ATT_WAVES][16];
  __shared__ float sCr[ATT_WAVES][16];

  int b    = blockIdx.y;
  int wave = threadIdx.x >> 5;
  int lane = threadIdx.x & 31;
  int g    = lane >> 4;
  int ln   = lane & 15;
  int qbase = (blockIdx.x * ATT_WAVES + wave) * 16;

  const bf16* qrow = qT + ((size_t)b * HW + qbase) * CA;
  v16bf aQ0 = load_frag_a(qrow, CA);        // d = 0..31
  v16bf aQ1 = load_frag_a(qrow + 32, CA);   // d = 32..63

  v8f zero = {0.f, 0.f, 0.f, 0.f, 0.f, 0.f, 0.f, 0.f};
  v8f O[4] = {zero, zero, zero, zero};
  if (lane < 16) { sM[wave][ln] = -1e30f; sL[wave][ln] = 0.f; }

  const uint4* gK = reinterpret_cast<const uint4*>(kT + (size_t)b * HW * CA);
  const uint4* gV = reinterpret_cast<const uint4*>(vT + (size_t)b * HW * CA);
  uint4* dK = reinterpret_cast<uint4*>(sK);
  uint4* dV = reinterpret_cast<uint4*>(sV);

  for (int kb = 0; kb < HW; kb += 32) {
    __syncthreads();   // everyone done with previous sK/sV
    int base = (kb * CA) >> 3;               // uint4 index
#if HAVE_ASYNC_LDS
    // CDNA5 async DMA path: global -> LDS without bouncing through VGPRs.
    {
      typedef __attribute__((address_space(1))) v4i glob_v4i;
      typedef __attribute__((address_space(3))) v4i lds_v4i;
      for (int i = threadIdx.x; i < 256; i += 128) {
        __builtin_amdgcn_global_load_async_to_lds_b128(
            (glob_v4i*)(gK + base + i), (lds_v4i*)(dK + i), 0, 0);
        __builtin_amdgcn_global_load_async_to_lds_b128(
            (glob_v4i*)(gV + base + i), (lds_v4i*)(dV + i), 0, 0);
      }
      asm volatile("s_wait_asynccnt 0" ::: "memory");
    }
#else
    for (int i = threadIdx.x; i < 256; i += 128) {
      dK[i] = gK[base + i];
      dV[i] = gV[base + i];
    }
#endif
    if (kb + 32 < HW) {                       // hint next tile toward L2
      __builtin_prefetch(gK + base + 256 + (threadIdx.x & 15) * 16, 0, 1);
      __builtin_prefetch(gV + base + 256 + (threadIdx.x & 15) * 16, 0, 1);
    }
    __syncthreads();

    // S(16x32) = Q(16x64) * K_tile^T ; B[k=d][n=key] = sK[key*64 + d]
    v8f s0 = zero, s1 = zero;
    s0 = wmma_bf16(aQ0, load_frag_b(sK, 1, 64), s0);
    s0 = wmma_bf16(aQ1, load_frag_b(sK + 32, 1, 64), s0);
    s1 = wmma_bf16(aQ0, load_frag_b(sK + 16 * 64, 1, 64), s1);
    s1 = wmma_bf16(aQ1, load_frag_b(sK + 16 * 64 + 32, 1, 64), s1);

#pragma unroll
    for (int r = 0; r < 8; ++r) {
      int row = r + 8 * g;
      sS[wave][row * 32 + ln]      = s0[r];
      sS[wave][row * 32 + 16 + ln] = s1[r];
    }

    // online softmax, one row per lane (lanes 0..15); wave-local LDS, in-order
    if (lane < 16) {
      float mo = sM[wave][ln];
      float mx = mo;
#pragma unroll
      for (int j = 0; j < 32; ++j) mx = fmaxf(mx, sS[wave][ln * 32 + j]);
      float corr = __expf(mo - mx);
      float sum = 0.f;
#pragma unroll
      for (int j = 0; j < 32; ++j) {
        float p = __expf(sS[wave][ln * 32 + j] - mx);
        sP[wave][ln * 32 + j] = (bf16)p;
        sum += p;
      }
      sL[wave][ln] = sL[wave][ln] * corr + sum;
      sM[wave][ln] = mx;
      sCr[wave][ln] = corr;
    }

    // rescale accumulator, then O(16x64) += P(16x32) * V(32x64)
#pragma unroll
    for (int r = 0; r < 8; ++r) {
      float cr = sCr[wave][r + 8 * g];
      O[0][r] *= cr; O[1][r] *= cr; O[2][r] *= cr; O[3][r] *= cr;
    }
    v16bf aP = load_frag_a(sP[wave], 32);
#pragma unroll
    for (int t = 0; t < 4; ++t)
      O[t] = wmma_bf16(aP, load_frag_b(sV + t * 16, 64, 1), O[t]);
  }

  // epilogue: divide by row sums, write out1[b][c][q]
#pragma unroll
  for (int t = 0; t < 4; ++t) {
#pragma unroll
    for (int r = 0; r < 8; ++r) {
      int row = r + 8 * g;
      float invl = 1.f / sL[wave][row];
      int c = t * 16 + ln;
      out1[((size_t)b * CA + c) * HW + qbase + row] = O[t][r] * invl;
    }
  }
}

// ---------------------------------------------------------------------------
// GEMM c1: y = w_c1(256x256) * concat(out1, x2)  -> BN -> SiLU -> bf16 ybuf
// ---------------------------------------------------------------------------
__global__ void __launch_bounds__(256) gemm_c1_kernel(
    const float* __restrict__ out1, const float* __restrict__ x,
    const float* __restrict__ w_c1,
    const float* __restrict__ bn_g, const float* __restrict__ bn_b,
    const float* __restrict__ bn_m, const float* __restrict__ bn_var,
    bf16* __restrict__ ybuf) {
  __shared__ bf16 sA[128 * 32];
  __shared__ bf16 sB[32 * 64];
  int b  = blockIdx.z;
  int m0 = blockIdx.y * 128;
  int n0 = blockIdx.x * 64;
  int wave = threadIdx.x >> 5;
  int lane = threadIdx.x & 31;
  int g = lane >> 4, ln = lane & 15;
  v8f zero = {0.f, 0.f, 0.f, 0.f, 0.f, 0.f, 0.f, 0.f};
  v8f acc[4] = {zero, zero, zero, zero};

  for (int k0 = 0; k0 < C; k0 += 32) {
    __syncthreads();
    for (int i = threadIdx.x; i < 128 * 32; i += 256) {
      int row = i >> 5, col = i & 31;
      sA[i] = (bf16)w_c1[(size_t)(m0 + row) * C + k0 + col];
    }
    for (int i = threadIdx.x; i < 32 * 64; i += 256) {
      int k = i >> 6, n = i & 63;
      int ci = k0 + k;
      float v = (ci < CA) ? out1[((size_t)b * CA + ci) * HW + n0 + n]
                          : x[((size_t)b * C + ci) * HW + n0 + n];
      sB[i] = (bf16)v;
    }
    __syncthreads();
    v16bf a = load_frag_a(sA + wave * 16 * 32, 32);
#pragma unroll
    for (int t = 0; t < 4; ++t)
      acc[t] = wmma_bf16(a, load_frag_b(sB + t * 16, 64, 1), acc[t]);
  }

#pragma unroll
  for (int t = 0; t < 4; ++t) {
#pragma unroll
    for (int r = 0; r < 8; ++r) {
      int co = m0 + wave * 16 + r + 8 * g;
      int n  = n0 + t * 16 + ln;
      float y = acc[t][r];
      float inv = rsqrtf(bn_var[co] + 1e-5f);
      y = (y - bn_m[co]) * (bn_g[co] * inv) + bn_b[co];
      y = y / (1.f + __expf(-y));              // SiLU
      ybuf[((size_t)b * C + co) * HW + n] = (bf16)y;
    }
  }
}

// ---------------------------------------------------------------------------
// GEMM c2: out = w_c2(256x256) * ybuf + b_c2   (f32 out)
// ---------------------------------------------------------------------------
__global__ void __launch_bounds__(256) gemm_c2_kernel(
    const bf16* __restrict__ ybuf, const float* __restrict__ w_c2,
    const float* __restrict__ b_c2, float* __restrict__ out) {
  __shared__ bf16 sA[128 * 32];
  __shared__ bf16 sB[32 * 64];
  int b  = blockIdx.z;
  int m0 = blockIdx.y * 128;
  int n0 = blockIdx.x * 64;
  int wave = threadIdx.x >> 5;
  int lane = threadIdx.x & 31;
  int g = lane >> 4, ln = lane & 15;
  v8f zero = {0.f, 0.f, 0.f, 0.f, 0.f, 0.f, 0.f, 0.f};
  v8f acc[4] = {zero, zero, zero, zero};

  for (int k0 = 0; k0 < C; k0 += 32) {
    __syncthreads();
    for (int i = threadIdx.x; i < 128 * 32; i += 256) {
      int row = i >> 5, col = i & 31;
      sA[i] = (bf16)w_c2[(size_t)(m0 + row) * C + k0 + col];
    }
    for (int i = threadIdx.x; i < 32 * 64; i += 256) {
      int k = i >> 6, n = i & 63;
      sB[i] = ybuf[((size_t)b * C + k0 + k) * HW + n0 + n];
    }
    __syncthreads();
    v16bf a = load_frag_a(sA + wave * 16 * 32, 32);
#pragma unroll
    for (int t = 0; t < 4; ++t)
      acc[t] = wmma_bf16(a, load_frag_b(sB + t * 16, 64, 1), acc[t]);
  }

#pragma unroll
  for (int t = 0; t < 4; ++t) {
#pragma unroll
    for (int r = 0; r < 8; ++r) {
      int co = m0 + wave * 16 + r + 8 * g;
      int n  = n0 + t * 16 + ln;
      out[((size_t)b * C + co) * HW + n] = acc[t][r] + b_c2[co];
    }
  }
}

// ---------------------------------------------------------------------------
extern "C" void kernel_launch(void* const* d_in, const int* in_sizes, int n_in,
                              void* d_out, int out_size, void* d_ws, size_t ws_size,
                              hipStream_t stream) {
  const float* x      = (const float*)d_in[0];
  const float* w_dw1  = (const float*)d_in[1];
  const float* b_dw1  = (const float*)d_in[2];
  const float* w_dw2  = (const float*)d_in[3];
  const float* b_dw2  = (const float*)d_in[4];
  const float* w_k    = (const float*)d_in[5];
  const float* b_k    = (const float*)d_in[6];
  const float* w_v    = (const float*)d_in[7];
  const float* b_v    = (const float*)d_in[8];
  const float* w_c1   = (const float*)d_in[9];
  const float* bn_g   = (const float*)d_in[10];
  const float* bn_b   = (const float*)d_in[11];
  const float* bn_m   = (const float*)d_in[12];
  const float* bn_var = (const float*)d_in[13];
  const float* w_c2   = (const float*)d_in[14];
  const float* b_c2   = (const float*)d_in[15];
  float* out = (float*)d_out;

  // workspace layout (all fully overwritten before being read, each call)
  char* ws    = (char*)d_ws;
  bf16*  qT   = (bf16*)(ws + (size_t)0);           //  4 MB  (B,HW,64) bf16
  bf16*  kT   = (bf16*)(ws + ((size_t)4  << 20));  //  4 MB
  bf16*  vT   = (bf16*)(ws + ((size_t)8  << 20));  //  4 MB
  float* out1 = (float*)(ws + ((size_t)12 << 20)); //  8 MB  (B,64,HW) f32
  bf16*  ybuf = (bf16*)(ws + ((size_t)20 << 20));  // 16 MB  (B,256,HW) bf16
  bf16*  wb7  = (bf16*)(ws + ((size_t)36 << 20));  // 196 KB packed 7x7 w
  bf16*  wb11 = (bf16*)(ws + ((size_t)36 << 20) + (512 << 10)); // 484 KB

  // 0) pre-pack conv weights to bf16 [tap][co][ci]
  prepack_w_kernel<7><<<dim3((32 * CA * 49 + 255) / 256), 256, 0, stream>>>(
      w_dw1, wb7);
  prepack_w_kernel<11><<<dim3((32 * CA * 121 + 255) / 256), 256, 0, stream>>>(
      w_dw2, wb11);

  // 1) build Q via implicit-GEMM WMMA convs; K/V scalar 1x1
  conv_q_wmma_kernel<7, 3><<<dim3(H, B), 128, 0, stream>>>(
      x, wb7, b_dw1, qT, 0);
  conv_q_wmma_kernel<11, 5><<<dim3(H, B), 128, 0, stream>>>(
      x, wb11, b_dw2, qT, 32);
  kv_kernel<<<dim3(B * 64 * HW / 256), 256, 0, stream>>>(
      x, w_k, b_k, w_v, b_v, kT, vT);

  // 2) fused flash attention (bf16 WMMA) -> out1 f32
  attn_kernel<<<dim3(HW / (16 * ATT_WAVES), B), 128, 0, stream>>>(
      qT, kT, vT, out1);

  // 3) c1 GEMM + BN + SiLU -> bf16 ybuf
  gemm_c1_kernel<<<dim3(HW / 64, C / 128, B), 256, 0, stream>>>(
      out1, x, w_c1, bn_g, bn_b, bn_m, bn_var, ybuf);

  // 4) c2 GEMM + bias -> f32 out
  gemm_c2_kernel<<<dim3(HW / 64, C / 128, B), 256, 0, stream>>>(
      ybuf, w_c2, b_c2, out);
}
